// RNN_Build_43774306681348
// MI455X (gfx1250) — compile-verified, matrix operand order checked by reading, not run on hardware
//
#include <hip/hip_runtime.h>
#include <hip/hip_bf16.h>

// Shapes from the reference
#define B_ 1024
#define T_ 256
#define E_ 512
#define U_ 1024

// 32 batch rows per workgroup: each B-fragment load feeds 2 WMMAs (2 M-tiles),
// halving L2 weight traffic vs M=16.
#define MROWS 32

typedef __attribute__((ext_vector_type(16))) __bf16 v16bf;
typedef __attribute__((ext_vector_type(8)))  float  v8f;

// Packed-weight (bf16) offsets inside d_ws, in elements.
// Block = one 32(K)x16(N) WMMA B-fragment: 32 lanes x 16 bf16, lane-major (512 elems, 1KB).
// Matrix layout: [ntile][ktile][lane][16]
#define W0P_OFF 0
#define U0P_OFF (64 * 16 * 512)            // W0: 64 ntiles x 16 ktiles
#define W1P_OFF (U0P_OFF + 64 * 32 * 512)  // U0: 64 x 32
#define U1P_OFF (W1P_OFF + 64 * 32 * 512)  // W1: 64 x 32
#define PACK_TOTAL (U1P_OFF + 64 * 32 * 512)

// LDS row strides (bf16 elems), padded so 16-lane row accesses spread banks
#define XSTR 520
#define HSTR 1032
#define SH_X_BYTES  (MROWS * XSTR * 2)
#define SH_H_BYTES  (MROWS * HSTR * 2)
#define SMEM_BYTES  (SH_X_BYTES + 2 * SH_H_BYTES)   // ~161.5 KB < 320 KB/WGP

#if __has_builtin(__builtin_amdgcn_tanhf)
#define FAST_TANH(x) __builtin_amdgcn_tanhf(x)
#else
#define FAST_TANH(x) tanhf(x)
#endif

union FragB { uint4 u4[2]; v16bf v; };

// ---------------------------------------------------------------------------
// One-time pack: fp32 row-major [K,N] weights -> bf16 WMMA B-fragment layout.
// Per ISA 7.12.2 (16-bit 32x16 B): lanes 0-15 = N, hold K {0..7,16..23};
// lanes 16-31 = N, hold K {8..15,24..31}; two K values per VGPR slot.
// ---------------------------------------------------------------------------
__global__ void pack_weights_bf16(const float* __restrict__ W0,
                                  const float* __restrict__ U0,
                                  const float* __restrict__ W1,
                                  const float* __restrict__ U1,
                                  __bf16* __restrict__ dst) {
  unsigned gid = blockIdx.x * blockDim.x + threadIdx.x;
  if (gid >= PACK_TOTAL) return;
  const float* src;
  unsigned rel, ktiles;
  if (gid < (unsigned)U0P_OFF)      { src = W0; rel = gid;            ktiles = 16; }
  else if (gid < (unsigned)W1P_OFF) { src = U0; rel = gid - U0P_OFF;  ktiles = 32; }
  else if (gid < (unsigned)U1P_OFF) { src = W1; rel = gid - W1P_OFF;  ktiles = 32; }
  else                              { src = U1; rel = gid - U1P_OFF;  ktiles = 32; }
  unsigned i    = rel & 15u;
  unsigned lane = (rel >> 4) & 31u;
  unsigned blk  = rel >> 9;
  unsigned kt   = blk % ktiles;
  unsigned nt   = blk / ktiles;
  unsigned n    = nt * 16 + (lane & 15u);
  unsigned half = lane >> 4;
  unsigned j    = i >> 1, pos = i & 1u;
  unsigned kl   = ((j & 4u) ? 16u : 0u) + (half ? 8u : 0u) + ((j & 3u) << 1) + pos;
  unsigned k    = kt * 32 + kl;
  dst[gid] = (__bf16)src[(size_t)k * U_ + n];
}

// A-fragment (16x32 bf16) from an LDS tile: lane m=lane%16, two contiguous 16B chunks.
__device__ __forceinline__ v16bf load_a(const __bf16* sh, int stride, int kt, int lane) {
  int m = lane & 15, half = lane >> 4;
  const __bf16* p = sh + m * stride + kt * 32 + half * 8;
  FragB f;
  f.u4[0] = *(const uint4*)(p);       // K chunk 0 (8 elems)
  f.u4[1] = *(const uint4*)(p + 16);  // K chunk 1 (8 elems)
  return f.v;
}

// B-fragment: 32 contiguous bytes per lane from the packed stream.
__device__ __forceinline__ v16bf load_b(const __bf16* g, int blk, int lane) {
  const __bf16* p = g + ((size_t)blk << 9) + lane * 16;
  FragB f;
  f.u4[0] = *(const uint4*)(p);
  f.u4[1] = *(const uint4*)(p + 8);
  return f.v;
}

// ---------------------------------------------------------------------------
// Persistent recurrent kernel: 32 workgroups x 32 batch rows, 8 waves each.
// Each wave owns 8 N-tiles x 2 M-tiles; B-fragments reused across M-tiles.
// ---------------------------------------------------------------------------
__launch_bounds__(256, 1)
__global__ void rnn_persistent(const int*   __restrict__ inp,    // [B,T]
                               const float* __restrict__ emb,    // [V,E]
                               const __bf16* __restrict__ wpack,
                               const float* __restrict__ b0v,
                               const float* __restrict__ b1v,
                               const float* __restrict__ Wout,
                               const float* __restrict__ bout,
                               float* __restrict__ out) {
  extern __shared__ __align__(16) unsigned char smem[];
  __bf16* shX  = (__bf16*)smem;                              // [MROWS][XSTR]
  __bf16* shH0 = (__bf16*)(smem + SH_X_BYTES);               // [MROWS][HSTR]
  __bf16* shH1 = (__bf16*)(smem + SH_X_BYTES + SH_H_BYTES);  // [MROWS][HSTR]

  const int tid  = threadIdx.x;
  const int wave = tid >> 5;
  const int lane = tid & 31;
  const int row0 = blockIdx.x * MROWS;

  // h0 = h1 = 0
  for (int i = tid; i < MROWS * HSTR; i += 256) {
    shH0[i] = (__bf16)0.0f;
    shH1[i] = (__bf16)0.0f;
  }

  // Bias per (ntile, lane) -- C-tile column index depends only on lane%16
  float bia0[8], bia1[8];
#pragma unroll
  for (int nt = 0; nt < 8; ++nt) {
    int n = (wave * 8 + nt) * 16 + (lane & 15);
    bia0[nt] = b0v[n];
    bia1[nt] = b1v[n];
  }
  __syncthreads();

  const __bf16* w0p = wpack + W0P_OFF;
  const __bf16* u0p = wpack + U0P_OFF;
  const __bf16* w1p = wpack + W1P_OFF;
  const __bf16* u1p = wpack + U1P_OFF;

  const int xr = tid >> 3;         // embedding gather: 8 threads per row (32 rows)
  const int xc = (tid & 7) * 64;   // 64 consecutive cols each

  for (int t = 0; t < T_; ++t) {
    // ---- gather x_t tile: emb[inputs[b,t]] -> bf16 in LDS ----
    {
      int vrow = inp[(row0 + xr) * T_ + t];
      const float* src = emb + (size_t)vrow * E_ + xc;
      __bf16* dstp = shX + xr * XSTR + xc;
#pragma unroll
      for (int q = 0; q < 16; ++q) {
        float4 f = *(const float4*)(src + q * 4);
        dstp[q * 4 + 0] = (__bf16)f.x;
        dstp[q * 4 + 1] = (__bf16)f.y;
        dstp[q * 4 + 2] = (__bf16)f.z;
        dstp[q * 4 + 3] = (__bf16)f.w;
      }
    }
    __syncthreads();

    // ---- layer 0: a0 = x@W0 + h0@U0 ----
    v8f acc0[8], acc1[8];
#pragma unroll
    for (int nt = 0; nt < 8; ++nt) {
      acc0[nt] = (v8f){0, 0, 0, 0, 0, 0, 0, 0};
      acc1[nt] = (v8f){0, 0, 0, 0, 0, 0, 0, 0};
    }

#pragma unroll 1
    for (int kt = 0; kt < E_ / 32; ++kt) {
      v16bf a0 = load_a(shX, XSTR, kt, lane);
      v16bf a1 = load_a(shX + 16 * XSTR, XSTR, kt, lane);
#pragma unroll
      for (int nt = 0; nt < 8; ++nt) {
        v16bf bb = load_b(w0p, (wave * 8 + nt) * (E_ / 32) + kt, lane);
        acc0[nt] = __builtin_amdgcn_wmma_f32_16x16x32_bf16(
            false, a0, false, bb, (short)0, acc0[nt], false, false);
        acc1[nt] = __builtin_amdgcn_wmma_f32_16x16x32_bf16(
            false, a1, false, bb, (short)0, acc1[nt], false, false);
      }
    }
#pragma unroll 1
    for (int kt = 0; kt < U_ / 32; ++kt) {
      v16bf a0 = load_a(shH0, HSTR, kt, lane);
      v16bf a1 = load_a(shH0 + 16 * HSTR, HSTR, kt, lane);
#pragma unroll
      for (int nt = 0; nt < 8; ++nt) {
        v16bf bb = load_b(u0p, (wave * 8 + nt) * (U_ / 32) + kt, lane);
        acc0[nt] = __builtin_amdgcn_wmma_f32_16x16x32_bf16(
            false, a0, false, bb, (short)0, acc0[nt], false, false);
        acc1[nt] = __builtin_amdgcn_wmma_f32_16x16x32_bf16(
            false, a1, false, bb, (short)0, acc1[nt], false, false);
      }
    }
#pragma unroll
    for (int nt = 0; nt < 8; ++nt)
#pragma unroll
      for (int r = 0; r < 8; ++r) {
        acc0[nt][r] = FAST_TANH(acc0[nt][r] + bia0[nt]);
        acc1[nt][r] = FAST_TANH(acc1[nt][r] + bia0[nt]);
      }

    __syncthreads();  // all waves finished reading old h0
    {
      int mbase = (lane >> 4) * 8;
      int nb = lane & 15;
#pragma unroll
      for (int nt = 0; nt < 8; ++nt) {
        int col = (wave * 8 + nt) * 16 + nb;
#pragma unroll
        for (int r = 0; r < 8; ++r) {
          shH0[(mbase + r) * HSTR + col]        = (__bf16)acc0[nt][r];
          shH0[(16 + mbase + r) * HSTR + col]   = (__bf16)acc1[nt][r];
        }
      }
    }
    __syncthreads();

    // ---- layer 1: a1 = h0@W1 + h1@U1 ----
#pragma unroll
    for (int nt = 0; nt < 8; ++nt) {
      acc0[nt] = (v8f){0, 0, 0, 0, 0, 0, 0, 0};
      acc1[nt] = (v8f){0, 0, 0, 0, 0, 0, 0, 0};
    }

#pragma unroll 1
    for (int kt = 0; kt < U_ / 32; ++kt) {
      v16bf a0 = load_a(shH0, HSTR, kt, lane);
      v16bf a1 = load_a(shH0 + 16 * HSTR, HSTR, kt, lane);
#pragma unroll
      for (int nt = 0; nt < 8; ++nt) {
        v16bf bb = load_b(w1p, (wave * 8 + nt) * (U_ / 32) + kt, lane);
        acc0[nt] = __builtin_amdgcn_wmma_f32_16x16x32_bf16(
            false, a0, false, bb, (short)0, acc0[nt], false, false);
        acc1[nt] = __builtin_amdgcn_wmma_f32_16x16x32_bf16(
            false, a1, false, bb, (short)0, acc1[nt], false, false);
      }
    }
#pragma unroll 1
    for (int kt = 0; kt < U_ / 32; ++kt) {
      v16bf a0 = load_a(shH1, HSTR, kt, lane);
      v16bf a1 = load_a(shH1 + 16 * HSTR, HSTR, kt, lane);
#pragma unroll
      for (int nt = 0; nt < 8; ++nt) {
        v16bf bb = load_b(u1p, (wave * 8 + nt) * (U_ / 32) + kt, lane);
        acc0[nt] = __builtin_amdgcn_wmma_f32_16x16x32_bf16(
            false, a0, false, bb, (short)0, acc0[nt], false, false);
        acc1[nt] = __builtin_amdgcn_wmma_f32_16x16x32_bf16(
            false, a1, false, bb, (short)0, acc1[nt], false, false);
      }
    }
#pragma unroll
    for (int nt = 0; nt < 8; ++nt)
#pragma unroll
      for (int r = 0; r < 8; ++r) {
        acc0[nt][r] = FAST_TANH(acc0[nt][r] + bia1[nt]);
        acc1[nt][r] = FAST_TANH(acc1[nt][r] + bia1[nt]);
      }

    __syncthreads();  // all waves finished reading old h1
    {
      int mbase = (lane >> 4) * 8;
      int nb = lane & 15;
#pragma unroll
      for (int nt = 0; nt < 8; ++nt) {
        int col = (wave * 8 + nt) * 16 + nb;
#pragma unroll
        for (int r = 0; r < 8; ++r) {
          shH1[(mbase + r) * HSTR + col]        = (__bf16)acc0[nt][r];
          shH1[(16 + mbase + r) * HSTR + col]   = (__bf16)acc1[nt][r];
        }
      }
    }
    __syncthreads();
  }

  // ---- epilogue: out[b] = sigmoid(h1[b] . Wout + bout) ----
  const float bo = bout[0];
#pragma unroll
  for (int rr = 0; rr < 4; ++rr) {
    int r = wave * 4 + rr;
    float s = 0.0f;
    for (int u = lane; u < U_; u += 32)
      s += (float)shH1[r * HSTR + u] * Wout[u];
#pragma unroll
    for (int m = 16; m >= 1; m >>= 1) s += __shfl_xor(s, m, 32);
    if (lane == 0) out[row0 + r] = 1.0f / (1.0f + __expf(-(s + bo)));
  }
}

extern "C" void kernel_launch(void* const* d_in, const int* in_sizes, int n_in,
                              void* d_out, int out_size, void* d_ws, size_t ws_size,
                              hipStream_t stream) {
  (void)in_sizes; (void)n_in; (void)out_size; (void)ws_size;
  const int*   inp  = (const int*)d_in[0];
  const float* emb  = (const float*)d_in[1];
  const float* W0   = (const float*)d_in[2];
  const float* U0   = (const float*)d_in[3];
  const float* b0   = (const float*)d_in[4];
  const float* W1   = (const float*)d_in[5];
  const float* U1   = (const float*)d_in[6];
  const float* b1   = (const float*)d_in[7];
  const float* Wout = (const float*)d_in[8];
  const float* bout = (const float*)d_in[9];
  float*  out   = (float*)d_out;
  __bf16* wpack = (__bf16*)d_ws;

  pack_weights_bf16<<<(PACK_TOTAL + 255) / 256, 256, 0, stream>>>(W0, U0, W1, U1, wpack);
  rnn_persistent<<<B_ / MROWS, 256, SMEM_BYTES, stream>>>(inp, emb, wpack, b0, b1,
                                                          Wout, bout, out);
}
